// MLAttention_89687507076046
// MI455X (gfx1250) — matrix operand order repaired and checked by reading, not compiled, not run on hardware
//
#include <hip/hip_runtime.h>
#include <hip/hip_bf16.h>
#include <math.h>

// ---------------------------------------------------------------------------
// MLAttention for MI455X (gfx1250, wave32, WMMA + async global->LDS).
//
//   scores[b,k,l] = sum_h X[b,l,h] * W[k,h]          (GEMM1, contraction H)
//   P = softmax_l(mask ? scores : -inf)
//   out[b,k,h]   = sum_l P[b,k,l] * X[b,l,h]         (GEMM2, contraction L)
//
// B=16 L=512 H=1024 K=8192.  ~275 GFLOP, ~0.6 GB compulsory HBM traffic
// (26 us @ 23.3 TB/s) -> balanced; use v_wmma_f32_16x16x32_f16 (f32 accum).
// X and W both fit in the 192 MB L2, so per-WG operand re-reads are L2 hits.
// Tile staging uses GLOBAL_LOAD_ASYNC_TO_LDS_B128 (ASYNCcnt): copies bypass
// the VGPR file and overlap the WMMA stream; the double-buffered loops need
// only  s_wait_asynccnt 0  before each publishing barrier.  B-fragments are
// explicitly register-double-buffered so WMMAs only wait on dscnt<=2.
// ---------------------------------------------------------------------------

#define B_  16
#define L_  512
#define H_  1024
#define K_  8192
#define KT  64          // K-rows per workgroup
#define NWAVES 16
#define NTHREADS 512

typedef __attribute__((ext_vector_type(16))) _Float16 v16h;
typedef __attribute__((ext_vector_type(8)))  _Float16 v8h;
typedef __attribute__((ext_vector_type(4)))  _Float16 v4h;
typedef __attribute__((ext_vector_type(8)))  float    v8f;

union V16 { v16h v; v8h h[2]; };

// LDS layout (bytes), total 160768 -> 2 WGs / 320KB WGP:
//   P      : 64 x 520 f16                     = 66560
//   Xs     : 2 x (512 x 40 f16)  (phase-1 B)  = 81920   (phase-2 XTs overlays)
//   Ws     : 2 x ( 64 x 40 f16)  (phase-1 A)  = 10240
//   red    : 2 x 4 x 64 f32                   =  2048
#define PSTRIDE 520           // f16 elems per P row (512 + 8 pad), 1040 B (16B aligned)
#define SSTRIDE 40            // f16 elems per staged row (32 + 8 pad), 80 B (16B aligned)
#define SMEM_BYTES (64*PSTRIDE*2 + 2*512*SSTRIDE*2 + 2*64*SSTRIDE*2 + 2*4*64*4)

// ---------------------------------------------------------------------------
// gfx1250 async global->LDS copy (16B per lane), tracked by ASYNCcnt.
// The DS address is the low 32 bits of the generic LDS pointer (aperture
// encoding: generic LDS addr = {SHARED_BASE[63:32], offset[31:0]}).
// NOTE: no "memory" clobber here -- the LDS writes are only observed after
// wait_async0() (volatile + memory clobber) and a barrier, and omitting it
// lets the scheduler keep ds_load/wmma software pipelining intact.
__device__ __forceinline__ void async_ld_b128(void* lds_dst, const void* gsrc) {
  unsigned off = (unsigned)(uintptr_t)lds_dst;
  unsigned long long ga = (unsigned long long)(uintptr_t)gsrc;
  asm volatile("global_load_async_to_lds_b128 %0, %1, off"
               :: "v"(off), "v"(ga));
}
__device__ __forceinline__ void wait_async0() {
  asm volatile("s_wait_asynccnt 0x0" ::: "memory");
}

// ---------------------------------------------------------------------------
// Prep kernel 1: f32 -> f16 (used for W).
__global__ void cvt_f16_kernel(const float* __restrict__ src,
                               _Float16* __restrict__ dst, int n4) {
  int i = blockIdx.x * blockDim.x + threadIdx.x;
  if (i >= n4) return;
  float4 f = ((const float4*)src)[i];
  v4h o = {(_Float16)f.x, (_Float16)f.y, (_Float16)f.z, (_Float16)f.w};
  ((v4h*)dst)[i] = o;
}

// Prep kernel 2: emit Xh[b][l][h] (f16) and XTh[b][h][l] (f16) in one pass.
__global__ void xpose_cvt_kernel(const float* __restrict__ X,
                                 _Float16* __restrict__ Xh,
                                 _Float16* __restrict__ XTh) {
  __shared__ float tile[32][33];
  int b  = blockIdx.z;
  int h0 = blockIdx.x * 32;
  int l0 = blockIdx.y * 32;
  const float* src = X + ((size_t)b * L_ + l0) * H_ + h0;
#pragma unroll
  for (int i = 0; i < 4; ++i) {
    int l = threadIdx.y + i * 8;
    float v = src[(size_t)l * H_ + threadIdx.x];
    tile[l][threadIdx.x] = v;
    Xh[((size_t)b * L_ + l0 + l) * H_ + h0 + threadIdx.x] = (_Float16)v;
  }
  __syncthreads();
#pragma unroll
  for (int i = 0; i < 4; ++i) {
    int h = threadIdx.y + i * 8;
    XTh[((size_t)b * H_ + h0 + h) * L_ + l0 + threadIdx.x] =
        (_Float16)tile[threadIdx.x][h];
  }
}

// ---------------------------------------------------------------------------
// Staging: async 16B copies into padded LDS rows (all 16B aligned).
__device__ __forceinline__ void stage_x(const _Float16* gXb, _Float16* dst,
                                        int h0, int tid) {
  // 512 threads <-> 512 L rows, 32 f16 (4 x b128) each.
  const _Float16* s = gXb + (size_t)tid * H_ + h0;
  _Float16* d = dst + tid * SSTRIDE;
#pragma unroll
  for (int c = 0; c < 4; ++c) async_ld_b128(d + c * 8, s + c * 8);
}
__device__ __forceinline__ void stage_w(const _Float16* gW, _Float16* dst,
                                        int k0, int h0, int tid) {
  if (tid < 256) {
    int row = tid & 63, c = tid >> 6;
    async_ld_b128(dst + row * SSTRIDE + c * 8,
                  gW + (size_t)(k0 + row) * H_ + h0 + c * 8);
  }
}
__device__ __forceinline__ void stage_xt(const _Float16* gXTb, _Float16* dst,
                                         int HB, int l0, int tid) {
  // 512 H rows x 32 f16 = 2048 b128 tasks over 512 threads.
#pragma unroll
  for (int j = 0; j < 4; ++j) {
    int t = tid + j * NTHREADS;
    int row = t & 511, c = t >> 9;
    async_ld_b128(dst + row * SSTRIDE + c * 8,
                  gXTb + (size_t)(HB + row) * L_ + l0 + c * 8);
  }
}

// A fragment (16x32 f16, M x Kdim) from row-major LDS [row][32] :
//   lanes 0-15 : M=lane,   K = {0..7, 16..23}
//   lanes16-31 : M=lane-16,K = {8..15,24..31}
__device__ __forceinline__ v16h load_a(const _Float16* rowbase, int kbh) {
  V16 a;
  a.h[0] = *(const v8h*)(rowbase + kbh);
  a.h[1] = *(const v8h*)(rowbase + 16 + kbh);
  return a.v;
}
// B fragment (32x16 f16, Kdim x N) from N-major LDS [col][32] :
//   lanes 0-15 : N=lane,   K = {0..7, 16..23} ; lanes 16-31 : N=lane-16, K = {8..15, 24..31}
__device__ __forceinline__ v16h load_b(const _Float16* colbase, int laneHalf) {
  V16 b;
  b.h[0] = *(const v8h*)(colbase + laneHalf * 8);
  b.h[1] = *(const v8h*)(colbase + 16 + laneHalf * 8);
  return b.v;
}

// ---------------------------------------------------------------------------
__global__ __launch_bounds__(NTHREADS)
void mlattn_kernel(const _Float16* __restrict__ Xh,
                   const _Float16* __restrict__ XTh,
                   const _Float16* __restrict__ Wh,
                   const unsigned char* __restrict__ masks,
                   float* __restrict__ out) {
  extern __shared__ char smem[];
  _Float16* Pbuf = (_Float16*)smem;                         // 64 x 520
  _Float16* Xs   = Pbuf + 64 * PSTRIDE;                     // 2 x 512 x 40
  _Float16* Ws   = Xs + 2 * 512 * SSTRIDE;                  // 2 x  64 x 40
  float*    redA = (float*)(Ws + 2 * 64 * SSTRIDE);         // 4 x 64
  float*    redB = redA + 256;                              // 4 x 64

  const int tid = threadIdx.x;
  const int wave = tid >> 5, lane = tid & 31;
  const int lane15 = lane & 15, laneHalf = lane >> 4;
  const int b = blockIdx.y, k0 = blockIdx.x * KT;

  const _Float16* gX  = Xh  + (size_t)b * L_ * H_;
  const _Float16* gXT = XTh + (size_t)b * H_ * L_;
  const unsigned char* mk = masks + (size_t)b * L_;

  // ================= Phase 1: S = W_tile @ X^T  (64 x 512) ==================
  // wave -> (mt: 16 K-rows, nq: 128 L-cols = 8 N-tiles)
  const int mt = wave >> 2, nq = wave & 3;
  v8f acc[8] = {};

  int cur = 0;
  stage_x(gX, Xs, 0, tid);
  stage_w(Wh, Ws, k0, 0, tid);
  wait_async0();
  __syncthreads();
  for (int h0 = 0; h0 < H_; h0 += 32) {
    int nxt = cur ^ 1;
    if (h0 + 32 < H_) {
      stage_x(gX, Xs + nxt * 512 * SSTRIDE, h0 + 32, tid);
      stage_w(Wh, Ws + nxt * 64 * SSTRIDE, k0, h0 + 32, tid);
    }
    const _Float16* ws = Ws + cur * 64 * SSTRIDE;
    const _Float16* xs = Xs + cur * 512 * SSTRIDE;
    v16h a = load_a(ws + (mt * 16 + lane15) * SSTRIDE, laneHalf * 8);
    v16h bc = load_b(xs + (nq * 128 + lane15) * SSTRIDE, laneHalf);
#pragma unroll
    for (int t = 0; t < 8; ++t) {
      v16h bn = bc;
      if (t < 7)
        bn = load_b(xs + (nq * 128 + (t + 1) * 16 + lane15) * SSTRIDE, laneHalf);
      acc[t] = __builtin_amdgcn_wmma_f32_16x16x32_f16(
          false, a, false, bc, (short)0, acc[t], false, false);
      bc = bn;
    }
    wait_async0();   // next-buffer async copies have landed in LDS
    __syncthreads();
    cur = nxt;
  }

  // ================= Softmax over L (masked) ================================
  // C layout: VGPR v of lane -> row (v + 8*laneHalf), col = tile*16 + lane15.
  int mbits = 0;
#pragma unroll
  for (int t = 0; t < 8; ++t)
    if (mk[nq * 128 + t * 16 + lane15]) mbits |= 1 << t;

  float m8[8];
#pragma unroll
  for (int v = 0; v < 8; ++v) {
    float m = -INFINITY;
#pragma unroll
    for (int t = 0; t < 8; ++t)
      if ((mbits >> t) & 1) m = fmaxf(m, acc[t][v]);
    m = fmaxf(m, __shfl_xor(m, 8, 32));
    m = fmaxf(m, __shfl_xor(m, 4, 32));
    m = fmaxf(m, __shfl_xor(m, 2, 32));
    m = fmaxf(m, __shfl_xor(m, 1, 32));
    m8[v] = m;
  }
  if (lane15 == 0) {
#pragma unroll
    for (int v = 0; v < 8; ++v)
      redA[nq * 64 + mt * 16 + laneHalf * 8 + v] = m8[v];
  }
  __syncthreads();
  float rmax[8];
#pragma unroll
  for (int v = 0; v < 8; ++v) {
    int idx = mt * 16 + laneHalf * 8 + v;
    float m = redA[idx];
    m = fmaxf(m, redA[64 + idx]);
    m = fmaxf(m, redA[128 + idx]);
    m = fmaxf(m, redA[192 + idx]);
    rmax[v] = m;
  }
  float s8[8];
#pragma unroll
  for (int v = 0; v < 8; ++v) {
    float s = 0.f;
#pragma unroll
    for (int t = 0; t < 8; ++t) {
      float e = ((mbits >> t) & 1) ? __expf(acc[t][v] - rmax[v]) : 0.f;
      acc[t][v] = e;
      s += e;
    }
    s += __shfl_xor(s, 8, 32);
    s += __shfl_xor(s, 4, 32);
    s += __shfl_xor(s, 2, 32);
    s += __shfl_xor(s, 1, 32);
    s8[v] = s;
  }
  if (lane15 == 0) {
#pragma unroll
    for (int v = 0; v < 8; ++v)
      redB[nq * 64 + mt * 16 + laneHalf * 8 + v] = s8[v];
  }
  __syncthreads();
#pragma unroll
  for (int v = 0; v < 8; ++v) {
    int idx = mt * 16 + laneHalf * 8 + v;
    float inv = 1.f / (redB[idx] + redB[64 + idx] + redB[128 + idx] + redB[192 + idx]);
    int row = mt * 16 + laneHalf * 8 + v;
#pragma unroll
    for (int t = 0; t < 8; ++t) {
      int col = nq * 128 + t * 16 + lane15;
      Pbuf[row * PSTRIDE + col] = (_Float16)(acc[t][v] * inv);
    }
  }
  __syncthreads();

  // ================= Phase 2: out_tile = P @ X  (64 x 1024) =================
  // wave -> (mt2: 16 K-rows, nq2: 128 H-cols = 8 N-tiles) per 512-wide block.
  const int mt2 = mt, nq2 = nq;
  _Float16* XTs = Xs;  // overlay phase-1 staging region (2 x 512 x 40)
  for (int hb = 0; hb < 2; ++hb) {
    const int HB = hb * 512;
    v8f acc2[8] = {};
    cur = 0;
    stage_xt(gXT, XTs, HB, 0, tid);
    wait_async0();
    __syncthreads();
    for (int l0 = 0; l0 < L_; l0 += 32) {
      int nxt = cur ^ 1;
      if (l0 + 32 < L_) stage_xt(gXT, XTs + nxt * 512 * SSTRIDE, HB, l0 + 32, tid);
      const _Float16* xts = XTs + cur * 512 * SSTRIDE;
      v16h a = load_a(Pbuf + (mt2 * 16 + lane15) * PSTRIDE + l0, laneHalf * 8);
      v16h bc = load_b(xts + (nq2 * 128 + lane15) * SSTRIDE, laneHalf);
#pragma unroll
      for (int t = 0; t < 8; ++t) {
        v16h bn = bc;
        if (t < 7)
          bn = load_b(xts + (nq2 * 128 + (t + 1) * 16 + lane15) * SSTRIDE,
                      laneHalf);
        acc2[t] = __builtin_amdgcn_wmma_f32_16x16x32_f16(
            false, a, false, bc, (short)0, acc2[t], false, false);
        bc = bn;
      }
      wait_async0();
      __syncthreads();
      cur = nxt;
    }
#pragma unroll
    for (int t = 0; t < 8; ++t) {
#pragma unroll
      for (int v = 0; v < 8; ++v) {
        int row = k0 + mt2 * 16 + laneHalf * 8 + v;
        int col = HB + nq2 * 128 + t * 16 + lane15;
        out[((size_t)b * K_ + row) * H_ + col] = acc2[t][v];
      }
    }
  }
}

// ---------------------------------------------------------------------------
extern "C" void kernel_launch(void* const* d_in, const int* in_sizes, int n_in,
                              void* d_out, int out_size, void* d_ws, size_t ws_size,
                              hipStream_t stream) {
  const float* X = (const float*)d_in[0];
  const unsigned char* masks = (const unsigned char*)d_in[1];  // jnp bool = 1B
  const float* W = (const float*)d_in[2];
  float* out = (float*)d_out;

  // Workspace: Xh | XTh | Wh  (3 x 8,388,608 f16 = 48 MB)
  _Float16* Xh  = (_Float16*)d_ws;
  _Float16* XTh = Xh + (size_t)B_ * L_ * H_;
  _Float16* Wh  = XTh + (size_t)B_ * L_ * H_;

  // W f32 -> f16
  {
    int n4 = (K_ * H_) / 4;
    cvt_f16_kernel<<<n4 / 256, 256, 0, stream>>>(W, Wh, n4);
  }
  // X f32 -> (Xh, XTh) f16
  xpose_cvt_kernel<<<dim3(H_ / 32, L_ / 32, B_), dim3(32, 8), 0, stream>>>(
      X, Xh, XTh);

  // Fused GEMM1 + softmax + GEMM2
  (void)hipFuncSetAttribute((const void*)mlattn_kernel,
                            hipFuncAttributeMaxDynamicSharedMemorySize,
                            SMEM_BYTES);
  mlattn_kernel<<<dim3(K_ / KT, B_), NTHREADS, SMEM_BYTES, stream>>>(
      Xh, XTh, Wh, masks, out);
  (void)in_sizes; (void)n_in; (void)out_size; (void)ws_size;
}
